// PolicyNetwork_42941083025678
// MI455X (gfx1250) — matrix operand order repaired and checked by reading, not compiled, not run on hardware
//
#include <hip/hip_runtime.h>
#include <hip/hip_bf16.h>

typedef __attribute__((ext_vector_type(16))) _Float16 v16h;
typedef __attribute__((ext_vector_type(8)))  _Float16 v8h;
typedef __attribute__((ext_vector_type(8)))  float    v8f;

#define NNODE 10000
#define NEDGE 160000
#define KEIG  16
#define FIN_  32
#define HID_  64
#define HEADS_ 4

// ---------------------------------------------------------------------------
// WMMA GEMM: C[M,Nout] = A[M,Kpad] (f16, row-major) * B (f16, stored as
// Bt[Nout,Kpad] row-major == B transposed) + bias[Nout].  One 16x16 output
// tile per wave, f32 accumulate via v_wmma_f32_16x16x32_f16.
// Fragment k order: contiguous, elem j -> K = 16*halfid + j (wave32).
// ---------------------------------------------------------------------------
__global__ void wmma_gemm_f16(const _Float16* __restrict__ A,
                              const _Float16* __restrict__ Bt,
                              const float* __restrict__ bias,
                              float* __restrict__ C,
                              int Ntiles, int Kpad, int Nout, int total_tiles) {
  int tile = blockIdx.x * (blockDim.x >> 5) + (threadIdx.x >> 5);
  if (tile >= total_tiles) return;               // wave-uniform branch
  int mtile = tile / Ntiles;
  int ntile = tile - mtile * Ntiles;
  int lane  = threadIdx.x & 31;
  int halfid = lane >> 4;
  int r     = lane & 15;
  const _Float16* Arow = A  + (size_t)(mtile * 16 + r) * Kpad;
  const _Float16* Brow = Bt + (size_t)(ntile * 16 + r) * Kpad;
  v8f acc = {};
  for (int k0 = 0; k0 < Kpad; k0 += 32) {
    // A frag: elem j -> K = k0 + halfid*16 + j, M = r  (contiguous in A row)
    v16h a = *(const v16h*)(Arow + k0 + halfid * 16);
    // B frag: elem j -> K = k0 + halfid*16 + j, N = r  (contiguous in Bt row)
    v16h b = *(const v16h*)(Brow + k0 + halfid * 16);
    acc = __builtin_amdgcn_wmma_f32_16x16x32_f16(false, a, false, b,
                                                 (short)0, acc, false, false);
  }
  float bv = bias[ntile * 16 + r];
  float* Cb = C + (size_t)(mtile * 16) * Nout + ntile * 16 + r;
#pragma unroll
  for (int v = 0; v < 8; v++)
    Cb[(size_t)(halfid * 8 + v) * Nout] = acc[v] + bv;   // M = halfid*8+v, N = r
}

// Repack fp32 W[fi,fo] (row-major) into f16 Wt[fo,Kpad] (transposed, zero-pad K)
__global__ void pack_wt(const float* __restrict__ W, const float* __restrict__ b,
                        int fi, int fo, int Kpad,
                        _Float16* __restrict__ Wt, float* __restrict__ bout) {
  int t = blockIdx.x * blockDim.x + threadIdx.x;
  if (t >= fo * Kpad) return;
  int c = t / Kpad, k = t - c * Kpad;
  float w = (k < fi) ? W[(size_t)k * fo + c] : 0.f;
  Wt[(size_t)c * Kpad + k] = (_Float16)w;
  if (k == 0) bout[c] = b[c];
}

// Layer-0 input: [node_features | eigenvecs[sub]] -> f16, K-padded
__global__ void build_x0(const float* __restrict__ nf, const float* __restrict__ eig,
                         const int* __restrict__ sub, _Float16* __restrict__ xh,
                         int n, int fin, int K, int Kpad) {
  int t = blockIdx.x * blockDim.x + threadIdx.x;
  if (t >= n * Kpad) return;
  int i = t / Kpad, k = t - i * Kpad;
  float v = 0.f;
  if (k < fin)          v = nf[(size_t)i * fin + k];
  else if (k < fin + K) v = eig[(size_t)sub[i] * K + (k - fin)];
  xh[t] = (_Float16)v;
}

__global__ void fill_u32(unsigned* p, unsigned v, int n) {
  int t = blockIdx.x * blockDim.x + threadIdx.x;
  if (t < n) p[t] = v;
}

__global__ void f32_to_f16(const float* __restrict__ a, _Float16* __restrict__ b, int n) {
  int t = blockIdx.x * blockDim.x + threadIdx.x;
  if (t < n) b[t] = (_Float16)a[t];
}

// attn accumulator init = skip branch (x@Ws + bs) already computed into QKVS S-section
__global__ void init_attn(const float* __restrict__ qkvs, float* __restrict__ attn,
                          int n, int HC) {
  int t = blockIdx.x * blockDim.x + threadIdx.x;
  if (t >= n * HC) return;
  int i = t / HC, j = t - i * HC;
  attn[t] = qkvs[(size_t)i * 4 * HC + 3 * HC + j];
}

// Pass A: logit[e,h] = <q[dst,h], k[src,h]> * scale; ordered-uint atomicMax per (dst,h)
__global__ void edge_logit_max(const int* __restrict__ ei, const float* __restrict__ qkvs,
                               float* __restrict__ logit, unsigned* __restrict__ mkey,
                               int E, int H, int ch, float scale) {
  int t = blockIdx.x * blockDim.x + threadIdx.x;
  if (t >= E * H) return;
  int e = t / H, h = t - e * H;
  int src = ei[e], dst = ei[E + e];
  int HC = H * ch, HC4 = 4 * HC;
  const float4* q = (const float4*)(qkvs + (size_t)dst * HC4 + h * ch);
  const float4* k = (const float4*)(qkvs + (size_t)src * HC4 + HC + h * ch);
  float acc = 0.f;
  for (int c = 0; c < (ch >> 2); c++) {
    float4 qa = q[c], ka = k[c];
    acc += qa.x * ka.x + qa.y * ka.y + qa.z * ka.z + qa.w * ka.w;
  }
  acc *= scale;
  logit[t] = acc;
  unsigned u = __float_as_uint(acc);
  unsigned key = (u & 0x80000000u) ? ~u : (u | 0x80000000u);
  atomicMax(mkey + (size_t)dst * H + h, key);
}

// Pass B: e = exp(logit - m[dst]); atomic segment-sum
__global__ void edge_exp_sum(const int* __restrict__ ei, const float* __restrict__ logit,
                             const unsigned* __restrict__ mkey, float* __restrict__ evals,
                             float* __restrict__ ssum, int E, int H) {
  int t = blockIdx.x * blockDim.x + threadIdx.x;
  if (t >= E * H) return;
  int e = t / H, h = t - e * H;
  int dst = ei[E + e];
  unsigned key = mkey[(size_t)dst * H + h];
  unsigned u = (key & 0x80000000u) ? (key & 0x7FFFFFFFu) : ~key;
  float m = __uint_as_float(u);
  float ev = __expf(logit[t] - m);
  evals[t] = ev;
  atomicAdd(ssum + (size_t)dst * H + h, ev);
}

// Pass C: attn[dst] += alpha * v[src]
__global__ void edge_accum(const int* __restrict__ ei, const float* __restrict__ qkvs,
                           const float* __restrict__ evals, const float* __restrict__ ssum,
                           float* __restrict__ attn, int E, int H, int ch) {
  int t = blockIdx.x * blockDim.x + threadIdx.x;
  if (t >= E * H) return;
  int e = t / H, h = t - e * H;
  int src = ei[e], dst = ei[E + e];
  int HC = H * ch, HC4 = 4 * HC;
  float alpha = evals[t] / ssum[(size_t)dst * H + h];
  const float* v = qkvs + (size_t)src * HC4 + 2 * HC + h * ch;
  float* o = attn + (size_t)dst * HC + h * ch;
  for (int c = 0; c < ch; c++) atomicAdd(o + c, alpha * v[c]);
}

// Per-channel mean/var over all nodes (axis-0 norm), one block per channel
__global__ void colstats(const float* __restrict__ h, int n, int C,
                         float* __restrict__ mu, float* __restrict__ rstd) {
  int c = blockIdx.x, t = threadIdx.x;
  float s = 0.f, s2 = 0.f;
  for (int i = t; i < n; i += blockDim.x) {
    float v = h[(size_t)i * C + c];
    s += v; s2 += v * v;
  }
  __shared__ float sh[256], sh2[256];
  sh[t] = s; sh2[t] = s2; __syncthreads();
  for (int o = 128; o > 0; o >>= 1) {
    if (t < o) { sh[t] += sh[t + o]; sh2[t] += sh2[t + o]; }
    __syncthreads();
  }
  if (t == 0) {
    float mean = sh[0] / (float)n;
    float var  = sh2[0] / (float)n - mean * mean;
    mu[c] = mean;
    rstd[c] = rsqrtf(var + 1e-5f);
  }
}

// normalize + affine + leaky, then concat eigen-pos, emit f16 K-padded next input
__global__ void norm_concat(const float* __restrict__ h, const float* __restrict__ mu,
                            const float* __restrict__ rstd, const float* __restrict__ g,
                            const float* __restrict__ be, const float* __restrict__ eig,
                            const int* __restrict__ sub, _Float16* __restrict__ xh,
                            int n, int C, int K, int Kpad) {
  int t = blockIdx.x * blockDim.x + threadIdx.x;
  if (t >= n * Kpad) return;
  int i = t / Kpad, k = t - i * Kpad;
  float v = 0.f;
  if (k < C) {
    float y = (h[(size_t)i * C + k] - mu[k]) * rstd[k] * g[k] + be[k];
    v = y > 0.f ? y : 0.01f * y;
  } else if (k < C + K) {
    v = eig[(size_t)sub[i] * K + (k - C)];
  }
  xh[t] = (_Float16)v;
}

// MLP head + log_softmax on the agent row (output: 8 floats)
__global__ void head_kernel(const float* __restrict__ xF, const int* __restrict__ agent,
                            const float* __restrict__ W0, const float* __restrict__ b0,
                            const float* __restrict__ W1, const float* __restrict__ b1,
                            float* __restrict__ out) {
  __shared__ float xa[64], h0[64], o8[8];
  int t = threadIdx.x;
  int ag = agent[0];
  xa[t] = xF[(size_t)ag * 64 + t];
  __syncthreads();
  float a = b0[t];
  for (int j = 0; j < 64; j++) a += xa[j] * W0[j * 64 + t];
  h0[t] = a > 0.f ? a : 0.01f * a;
  __syncthreads();
  if (t < 8) {
    float o = b1[t];
    for (int j = 0; j < 64; j++) o += h0[j] * W1[j * 8 + t];
    o8[t] = o;
  }
  __syncthreads();
  if (t == 0) {
    float m = o8[0];
    for (int i = 1; i < 8; i++) m = fmaxf(m, o8[i]);
    float s = 0.f;
    for (int i = 0; i < 8; i++) s += __expf(o8[i] - m);
    float lse = m + logf(s);
    for (int i = 0; i < 8; i++) out[i] = o8[i] - lse;
  }
}

extern "C" void kernel_launch(void* const* d_in, const int* in_sizes, int n_in,
                              void* d_out, int out_size, void* d_ws, size_t ws_size,
                              hipStream_t stream) {
  const int N = NNODE, E = NEDGE, Kd = KEIG, FIN = FIN_, HID = HID_, HEADS = HEADS_;
  const float* nf  = (const float*)d_in[0];
  const int*   ei  = (const int*)d_in[1];
  const int*   sub = (const int*)d_in[2];
  const float* eig = (const float*)d_in[3];
  // params dict flattened in sorted-key order (28+8 leaves):
  // per block: Wk,Wkb,Wq,Wqb,Ws,Wsb,Wv,Wvb,be,g,pW,pb  (uppercase sorts first)
  const float* P[36];
  for (int i = 0; i < 36; i++) P[i] = (const float*)d_in[4 + i];
  const int* agent = (const int*)d_in[40];

  char* ws = (char*)d_ws;
  size_t cur = 0;
  auto walloc = [&](size_t bytes) -> char* {
    char* p = ws + cur;
    cur += (bytes + 255) & ~(size_t)255;
    return p;
  };
  _Float16* xh    = (_Float16*)walloc((size_t)N * 96 * 2);
  _Float16* wt    = (_Float16*)walloc((size_t)1024 * 96 * 2);
  float*    bcat  = (float*)walloc(1024 * 4);
  float*    qkvs  = (float*)walloc((size_t)N * 1024 * 4);
  float*    logit = (float*)walloc((size_t)E * HEADS * 4);
  float*    evals = (float*)walloc((size_t)E * HEADS * 4);
  unsigned* mkey  = (unsigned*)walloc((size_t)N * HEADS * 4);
  float*    ssum  = (float*)walloc((size_t)N * HEADS * 4);
  float*    attn  = (float*)walloc((size_t)N * 256 * 4);
  _Float16* attnh = (_Float16*)walloc((size_t)N * 256 * 2);
  float*    hbuf  = (float*)walloc((size_t)N * 64 * 4);
  float*    mu    = (float*)walloc(64 * 4);
  float*    rstd  = (float*)walloc(64 * 4);

  auto cdiv = [](int a, int b) { return (a + b - 1) / b; };

  auto run_tconv = [&](int Kpad, int fi, int H, int ch,
                       const float* Wq, const float* Wqb, const float* Wk, const float* Wkb,
                       const float* Wv, const float* Wvb, const float* Ws, const float* Wsb) {
    int HC = H * ch, HC4 = 4 * HC;
    int pt = HC * Kpad;
    pack_wt<<<cdiv(pt, 256), 256, 0, stream>>>(Wq, Wqb, fi, HC, Kpad, wt + (size_t)0 * HC * Kpad, bcat + 0);
    pack_wt<<<cdiv(pt, 256), 256, 0, stream>>>(Wk, Wkb, fi, HC, Kpad, wt + (size_t)1 * HC * Kpad, bcat + HC);
    pack_wt<<<cdiv(pt, 256), 256, 0, stream>>>(Wv, Wvb, fi, HC, Kpad, wt + (size_t)2 * HC * Kpad, bcat + 2 * HC);
    pack_wt<<<cdiv(pt, 256), 256, 0, stream>>>(Ws, Wsb, fi, HC, Kpad, wt + (size_t)3 * HC * Kpad, bcat + 3 * HC);
    int Mtiles = N / 16, Ntiles = HC4 / 16, tt = Mtiles * Ntiles;
    wmma_gemm_f16<<<cdiv(tt, 4), 128, 0, stream>>>(xh, wt, bcat, qkvs, Ntiles, Kpad, HC4, tt);
    fill_u32<<<cdiv(N * H, 256), 256, 0, stream>>>(mkey, 0u, N * H);
    fill_u32<<<cdiv(N * H, 256), 256, 0, stream>>>((unsigned*)ssum, 0u, N * H);
    init_attn<<<cdiv(N * HC, 256), 256, 0, stream>>>(qkvs, attn, N, HC);
    int ET = E * H;
    edge_logit_max<<<cdiv(ET, 256), 256, 0, stream>>>(ei, qkvs, logit, mkey, E, H, ch, 0.125f);
    edge_exp_sum<<<cdiv(ET, 256), 256, 0, stream>>>(ei, logit, mkey, evals, ssum, E, H);
    edge_accum<<<cdiv(ET, 256), 256, 0, stream>>>(ei, qkvs, evals, ssum, attn, E, H, ch);
  };

  // layer-0 input
  build_x0<<<cdiv(N * 64, 256), 256, 0, stream>>>(nf, eig, sub, xh, N, FIN, Kd, 64);

  for (int li = 0; li < 2; li++) {
    const float* const* B = P + li * 12;
    int Kpad = (li == 0) ? 64 : 96;
    int fi   = (li == 0) ? (FIN + Kd) : (HID + Kd);
    // B order: Wk,Wkb,Wq,Wqb,Ws,Wsb,Wv,Wvb,be,g,pW,pb
    run_tconv(Kpad, fi, HEADS, HID,
              /*Wq*/ B[2], B[3], /*Wk*/ B[0], B[1],
              /*Wv*/ B[6], B[7], /*Ws*/ B[4], B[5]);
    // proj 256 -> 64 (WMMA)
    f32_to_f16<<<cdiv(N * 256, 256), 256, 0, stream>>>(attn, attnh, N * 256);
    pack_wt<<<cdiv(64 * 256, 256), 256, 0, stream>>>(B[10], B[11], 256, 64, 256, wt, bcat);
    int Mtiles = N / 16, Ntiles = 4, tt = Mtiles * Ntiles;
    wmma_gemm_f16<<<cdiv(tt, 4), 128, 0, stream>>>(attnh, wt, bcat, hbuf, Ntiles, 256, 64, tt);
    // axis-0 layernorm + leaky + concat pos -> next f16 input [N,96]
    colstats<<<64, 256, 0, stream>>>(hbuf, N, 64, mu, rstd);
    norm_concat<<<cdiv(N * 96, 256), 256, 0, stream>>>(hbuf, mu, rstd,
                                                       /*g*/ B[9], /*be*/ B[8],
                                                       eig, sub, xh, N, 64, Kd, 96);
  }

  // final tconv: 1 head, 64 channels, fi=80  (f params: Wk,Wkb,Wq,Wqb,Ws,Wsb,Wv,Wvb)
  run_tconv(96, 80, 1, 64,
            /*Wq*/ P[26], P[27], /*Wk*/ P[24], P[25],
            /*Wv*/ P[30], P[31], /*Ws*/ P[28], P[29]);

  // MLP head + log_softmax on agent row
  head_kernel<<<1, 64, 0, stream>>>(attn, agent, P[32], P[33], P[34], P[35], (float*)d_out);
}
